// SimpleSelfAttention_40965398069529
// MI455X (gfx1250) — matrix operand order repaired and checked by reading, not compiled, not run on hardware
//
#include <hip/hip_runtime.h>

// ---------------------------------------------------------------------------
// CDNA5 (gfx1250) bf16-WMMA self-attention:
//   q = query Wq^T ; k = key Wk^T ; vT = Wv value^T
//   S = q k^T ; P = softmax(S/sqrt(E)) ; ao = P vT^T ; out = ao Wo^T + bo
// Every matmul is C[m,n] = sum_k A'[m,k] * B'[n,k]  (both operands row-major
// with contiguous K) -> v_wmma_f32_16x16x32_bf16 fragment layout.
// Pure-bf16 GEMMs (QK^T, PV) stage tiles with the Tensor Data Mover
// (tensor_load_to_lds + s_wait_tensorcnt) into a double-buffered LDS tile.
// ---------------------------------------------------------------------------

typedef __attribute__((ext_vector_type(8)))  __bf16 v8bf;
typedef __attribute__((ext_vector_type(16))) __bf16 v16bf;
typedef __attribute__((ext_vector_type(8)))  float  v8f;
typedef __attribute__((ext_vector_type(4)))  unsigned int u32x4;
typedef __attribute__((ext_vector_type(4)))  int  i32x4;
typedef __attribute__((ext_vector_type(8)))  int  i32x8;

#if defined(__HIP_DEVICE_COMPILE__) && \
    __has_builtin(__builtin_amdgcn_tensor_load_to_lds) && \
    __has_builtin(__builtin_amdgcn_s_wait_tensorcnt)
#define HAVE_TDM 1
#else
#define HAVE_TDM 0
#endif

__device__ __forceinline__ __bf16 f2bf(float f) {
  union { float f; unsigned u; } x; x.f = f;
  unsigned r = x.u + 0x7FFFu + ((x.u >> 16) & 1u);   // round-to-nearest-even
  union { unsigned short s; __bf16 b; } y; y.s = (unsigned short)(r >> 16);
  return y.b;
}

#if HAVE_TDM
// Issue a TDM load of a 128-row x 32-col bf16 tile (row stride ldElems) from
// gptr into LDS at byte offset ldsOff, padding 16B after every 64B row so the
// LDS image has an 80B row stride (SA=40 bf16), matching the fragment reader.
__device__ __forceinline__ void tdm_tile_load(unsigned ldsOff, const void* gptr, int ldElems) {
  unsigned long long ga = (unsigned long long)gptr;
  u32x4 g0 = { 1u,                                   // count=1, is_restore=0
               ldsOff,                               // lds_addr
               (unsigned)(ga & 0xFFFFFFFFu),         // global_addr lo
               (unsigned)((ga >> 32) & 0x01FFFFFFu)  // global_addr hi (bits 56:32)
                 | 0x80000000u };                    // type=2 ("image")
  i32x8 g1 = { (int)((1u << 16)                      // data_size = 2B
                   | (1u << 20)                      // pad_enable
                   | (3u << 22)                      // pad_interval: 16 DWORDs (64B)
                   | (3u << 25)),                    // pad_amount:   4 DWORDs (16B)
               (int)(32u << 16),                     // tensor_dim0 = 32 (lo16)
               (int)(128u << 16),                    // tensor_dim0 hi=0 | tensor_dim1=128 (lo16)
               (int)(32u << 16),                     // tensor_dim1 hi=0 | tile_dim0=32
               128,                                  // tile_dim1=128, tile_dim2=0
               ldElems,                              // tensor_dim0_stride lo32
               0, 0 };                               // stride hi / dim1_stride (unused, 2D)
  i32x4 z4 = { 0, 0, 0, 0 };
#if __clang_major__ >= 23
  i32x8 z8 = { 0, 0, 0, 0, 0, 0, 0, 0 };
  __builtin_amdgcn_tensor_load_to_lds(g0, g1, z4, z4, z8, 0);
#else
  __builtin_amdgcn_tensor_load_to_lds(g0, g1, z4, z4, 0);
#endif
}
__device__ __forceinline__ void tdm_wait() {
  __builtin_amdgcn_s_wait_tensorcnt((short)0);
}
#else
__device__ __forceinline__ void tdm_tile_load(unsigned, const void*, int) {}
__device__ __forceinline__ void tdm_wait() {}
#endif

// Fragment loads + 8 WMMAs for one 32-wide K chunk from LDS tile pair.
__device__ __forceinline__ void mma_tiles(const __bf16* __restrict__ AsB,
                                          const __bf16* __restrict__ BsB,
                                          int wm, int wn, int l16, int h,
                                          v8f (&acc)[2][4])
{
  constexpr int SA = 40;
  v16bf af[2];
#pragma unroll
  for (int ms = 0; ms < 2; ++ms) {
    const int r = wm * 32 + ms * 16 + l16;
    v8bf lo = *(const v8bf*)&AsB[r * SA + h * 8];
    v8bf hi = *(const v8bf*)&AsB[r * SA + 16 + h * 8];
    af[ms] = __builtin_shufflevector(lo, hi, 0,1,2,3,4,5,6,7,8,9,10,11,12,13,14,15);
  }
  v16bf bfr[4];
#pragma unroll
  for (int ns = 0; ns < 4; ++ns) {
    const int c = wn * 64 + ns * 16 + l16;
    v8bf lo = *(const v8bf*)&BsB[c * SA + h * 8];
    v8bf hi = *(const v8bf*)&BsB[c * SA + 16 + h * 8];
    bfr[ns] = __builtin_shufflevector(lo, hi, 0,1,2,3,4,5,6,7,8,9,10,11,12,13,14,15);
  }
#pragma unroll
  for (int ms = 0; ms < 2; ++ms)
#pragma unroll
    for (int ns = 0; ns < 4; ++ns)
      acc[ms][ns] = __builtin_amdgcn_wmma_f32_16x16x32_bf16(
          false, af[ms], false, bfr[ns], (short)0, acc[ms][ns], false, false);
}

// ---------------------------------------------------------------------------
// Generic tiled GEMM, C = A' * B'^T.  Block tile 128x128, K in 32-chunks.
// 256 threads = 8 waves (4x2); wave tile 32x64 (2x4 WMMA accumulators).
// ASYNC (requires bf16 A and B): TDM double-buffered staging.
// ---------------------------------------------------------------------------
template <bool A_F32, bool B_F32, bool OUT_BF16, bool BIAS, bool ASYNC>
__global__ __launch_bounds__(256) void gemm_bt(
    const void* __restrict__ Ap, long long bsA, int lda,
    const void* __restrict__ Bp, long long bsB, int ldb,
    void*       __restrict__ Cp, long long bsC, int ldc,
    const float* __restrict__ bias, int K)
{
  constexpr int SA   = 40;          // padded LDS row stride (80B)
  constexpr int TILE = 128 * SA;
  constexpr bool kTDM = ASYNC && (HAVE_TDM != 0);
  __shared__ __bf16 As[(ASYNC ? 2 : 1) * TILE];
  __shared__ __bf16 Bs[(ASYNC ? 2 : 1) * TILE];

  const int t    = threadIdx.x;
  const int lane = t & 31;
  const int l16  = lane & 15;
  const int h    = lane >> 4;
  const int wave = t >> 5;
  const int wm   = wave & 3;
  const int wn   = wave >> 2;
  const int z    = blockIdx.z;

  const long long rowM0 = (long long)blockIdx.y * 128;
  const long long colN0 = (long long)blockIdx.x * 128;

  v8f acc[2][4];
#pragma unroll
  for (int i = 0; i < 2; ++i)
#pragma unroll
    for (int j = 0; j < 4; ++j) acc[i][j] = (v8f)(0.0f);

  if constexpr (kTDM) {
    // -------- TDM double-buffered pipeline (bf16 A and B) --------
    const __bf16* Abase = (const __bf16*)Ap + (long long)z * bsA + rowM0 * (long long)lda;
    const __bf16* Bbase = (const __bf16*)Bp + (long long)z * bsB + colN0 * (long long)ldb;
    const int NC = K >> 5;
    if (wave == 0) {
      tdm_tile_load((unsigned)(unsigned long long)&As[0], Abase, lda);
      tdm_tile_load((unsigned)(unsigned long long)&Bs[0], Bbase, ldb);
      tdm_wait();
    }
    __syncthreads();
    for (int i = 0; i < NC; ++i) {
      const int cur = i & 1;
      if (wave == 0 && i + 1 < NC) {   // stage next chunk while computing this one
        tdm_tile_load((unsigned)(unsigned long long)&As[(cur ^ 1) * TILE],
                      Abase + (long long)(i + 1) * 32, lda);
        tdm_tile_load((unsigned)(unsigned long long)&Bs[(cur ^ 1) * TILE],
                      Bbase + (long long)(i + 1) * 32, ldb);
      }
      mma_tiles(&As[cur * TILE], &Bs[cur * TILE], wm, wn, l16, h, acc);
      if (wave == 0) tdm_wait();       // next buffer landed
      __syncthreads();
    }
  } else {
    // -------- vector-load path (f32 sources converted to bf16 in flight) ----
    const int ldRow = t >> 1;          // 2 threads per row, 16 elems each
    const int ldCol = (t & 1) << 4;
    const float*  AfRow = (const float*) Ap + (long long)z * bsA + (rowM0 + ldRow) * (long long)lda + ldCol;
    const __bf16* AbRow = (const __bf16*)Ap + (long long)z * bsA + (rowM0 + ldRow) * (long long)lda + ldCol;
    const float*  BfRow = (const float*) Bp + (long long)z * bsB + (colN0 + ldRow) * (long long)ldb + ldCol;
    const __bf16* BbRow = (const __bf16*)Bp + (long long)z * bsB + (colN0 + ldRow) * (long long)ldb + ldCol;

    v8bf a0, a1, b0, b1;
    auto load_regs = [&](int k0) {
      if (A_F32) {
        const float4* p = (const float4*)(AfRow + k0);
        float fa[16];
#pragma unroll
        for (int i = 0; i < 4; ++i) {
          float4 f = p[i];
          fa[4*i+0] = f.x; fa[4*i+1] = f.y; fa[4*i+2] = f.z; fa[4*i+3] = f.w;
        }
#pragma unroll
        for (int i = 0; i < 8; ++i) { a0[i] = f2bf(fa[i]); a1[i] = f2bf(fa[8+i]); }
      } else {
        const v8bf* p = (const v8bf*)(AbRow + k0);
        a0 = p[0]; a1 = p[1];
      }
      if (B_F32) {
        const float4* p = (const float4*)(BfRow + k0);
        float fb[16];
#pragma unroll
        for (int i = 0; i < 4; ++i) {
          float4 f = p[i];
          fb[4*i+0] = f.x; fb[4*i+1] = f.y; fb[4*i+2] = f.z; fb[4*i+3] = f.w;
        }
#pragma unroll
        for (int i = 0; i < 8; ++i) { b0[i] = f2bf(fb[i]); b1[i] = f2bf(fb[8+i]); }
      } else {
        const v8bf* p = (const v8bf*)(BbRow + k0);
        b0 = p[0]; b1 = p[1];
      }
    };

    load_regs(0);                      // prologue
    for (int k0 = 0; k0 < K; k0 += 32) {
      __syncthreads();                 // previous chunk's compute done
      *(v8bf*)&As[ldRow * SA + ldCol]     = a0;
      *(v8bf*)&As[ldRow * SA + ldCol + 8] = a1;
      *(v8bf*)&Bs[ldRow * SA + ldCol]     = b0;
      *(v8bf*)&Bs[ldRow * SA + ldCol + 8] = b1;
      __syncthreads();                 // tiles visible
      if (k0 + 32 < K) load_regs(k0 + 32);   // prefetch next chunk over compute
      mma_tiles(As, Bs, wm, wn, l16, h, acc);
    }
  }

  // Epilogue. C layout: VGPR r <-> M = (lane>>4)*8 + r ; N = lane&15.
#pragma unroll
  for (int ms = 0; ms < 2; ++ms)
#pragma unroll
    for (int ns = 0; ns < 4; ++ns) {
      const long long row0 = rowM0 + wm * 32 + ms * 16 + h * 8;
      const long long col  = colN0 + wn * 64 + ns * 16 + l16;
      const float bb = BIAS ? bias[col] : 0.0f;
#pragma unroll
      for (int r = 0; r < 8; ++r) {
        const float vv = acc[ms][ns][r] + bb;
        const long long idx = (long long)z * bsC + (row0 + r) * (long long)ldc + col;
        if (OUT_BF16) ((__bf16*)Cp)[idx] = f2bf(vv);
        else          ((float*)  Cp)[idx] = vv;
      }
    }
}

// ---------------------------------------------------------------------------
// Row softmax: P[row,:] = softmax(S[row,:] * scale), f32 -> bf16.
// ---------------------------------------------------------------------------
__global__ __launch_bounds__(256) void softmax_rows(
    const void* __restrict__ Sp, void* __restrict__ Pp, int n, float scale)
{
  __shared__ float red[256];
  const long long row = blockIdx.x;
  const float* src = (const float*)Sp + row * (long long)n;
  __bf16*      dst = (__bf16*)Pp      + row * (long long)n;
  const int t = threadIdx.x;

  float x[8];
#pragma unroll
  for (int i = 0; i < 2; ++i) {
    float4 f = ((const float4*)(src + t * 8))[i];
    x[4*i+0] = f.x * scale; x[4*i+1] = f.y * scale;
    x[4*i+2] = f.z * scale; x[4*i+3] = f.w * scale;
  }
  float m = x[0];
#pragma unroll
  for (int i = 1; i < 8; ++i) m = fmaxf(m, x[i]);
  red[t] = m; __syncthreads();
  for (int s = 128; s > 0; s >>= 1) { if (t < s) red[t] = fmaxf(red[t], red[t + s]); __syncthreads(); }
  m = red[0]; __syncthreads();

  float sum = 0.0f;
#pragma unroll
  for (int i = 0; i < 8; ++i) { x[i] = __expf(x[i] - m); sum += x[i]; }
  red[t] = sum; __syncthreads();
  for (int s = 128; s > 0; s >>= 1) { if (t < s) red[t] += red[t + s]; __syncthreads(); }
  const float inv = 1.0f / red[0];
#pragma unroll
  for (int i = 0; i < 8; ++i) dst[t * 8 + i] = f2bf(x[i] * inv);
}

// ---------------------------------------------------------------------------
// Host orchestration
// ---------------------------------------------------------------------------
extern "C" void kernel_launch(void* const* d_in, const int* in_sizes, int n_in,
                              void* d_out, int out_size, void* d_ws, size_t ws_size,
                              hipStream_t stream)
{
  (void)in_sizes; (void)n_in; (void)out_size; (void)ws_size;
  const int B = 4, N = 2048, E = 512;
  const long long NE = (long long)N * E;

  const void* value = d_in[0];
  const void* key_  = d_in[1];
  const void* query = d_in[2];
  const void* Wq    = d_in[3];
  const void* Wk    = d_in[4];
  const void* Wv    = d_in[5];
  const void* Wo    = d_in[6];
  const float* bo   = (const float*)d_in[7];

  // ws: qb/kb/vT/ao bf16 (8 MiB each), S1 f32 (16 MiB), P1 bf16 (8 MiB)
  char* ws = (char*)d_ws;
  void* qb = ws;                                    // [B,N,E] bf16
  void* kb = ws + (size_t)8  * 1024 * 1024;         // [B,N,E] bf16
  void* vT = ws + (size_t)16 * 1024 * 1024;         // [B,E,N] bf16 (v transposed)
  void* ao = ws + (size_t)24 * 1024 * 1024;         // [B,N,E] bf16
  void* S1 = ws + (size_t)32 * 1024 * 1024;         // [N,N]   f32  (per-batch)
  void* P1 = ws + (size_t)48 * 1024 * 1024;         // [N,N]   bf16 (per-batch)

  dim3 blk(256, 1, 1);
  const float scale = 0.044194173824159216f;        // 1/sqrt(512)

  // q = query Wq^T   (M=8192, N=512, K=512)
  gemm_bt<true, true, true, false, false><<<dim3(E/128, (B*N)/128, 1), blk, 0, stream>>>(
      query, 0, E, Wq, 0, E, qb, 0, E, nullptr, E);
  // k = key Wk^T
  gemm_bt<true, true, true, false, false><<<dim3(E/128, (B*N)/128, 1), blk, 0, stream>>>(
      key_, 0, E, Wk, 0, E, kb, 0, E, nullptr, E);
  // vT[b] = Wv value[b]^T   (M=512, N=2048, K=512; batched)
  gemm_bt<true, true, true, false, false><<<dim3(N/128, E/128, B), blk, 0, stream>>>(
      Wv, 0, E, value, NE, E, vT, NE, N, nullptr, E);

  for (int b = 0; b < B; ++b) {
    const __bf16* qbB = (const __bf16*)qb + (long long)b * NE;
    const __bf16* kbB = (const __bf16*)kb + (long long)b * NE;
    const __bf16* vTB = (const __bf16*)vT + (long long)b * NE;
    __bf16*       aoB = (__bf16*)ao       + (long long)b * NE;

    // S = q k^T   (M=N=2048, K=512) -- TDM staged
    gemm_bt<false, false, false, false, true><<<dim3(N/128, N/128, 1), blk, 0, stream>>>(
        qbB, 0, E, kbB, 0, E, S1, 0, N, nullptr, E);
    // P = softmax(S * scale)
    softmax_rows<<<dim3(N, 1, 1), blk, 0, stream>>>(S1, P1, N, scale);
    // ao = P vT^T   (M=2048, N=512, K=2048) -- TDM staged
    gemm_bt<false, false, true, false, true><<<dim3(E/128, N/128, 1), blk, 0, stream>>>(
        P1, 0, N, vTB, 0, N, aoB, 0, E, nullptr, N);
  }

  // out = ao Wo^T + bo   (M=8192, N=512, K=512), f32 output
  gemm_bt<false, true, false, true, false><<<dim3(E/128, (B*N)/128, 1), blk, 0, stream>>>(
      ao, 0, E, Wo, 0, E, d_out, 0, E, bo, E);
}
